// CNN_DiSMo_29283087024838
// MI455X (gfx1250) — matrix operand order repaired and checked by reading, not compiled, not run on hardware
//
#include <hip/hip_runtime.h>
#include <hip/hip_bf16.h>

typedef __attribute__((ext_vector_type(16))) _Float16 v16h;
typedef __attribute__((ext_vector_type(8)))  _Float16 v8h;
typedef __attribute__((ext_vector_type(8)))  float    v8f;

#define QN 128
#define SN 25
#define TN 16
#define DN 2048
#define CHUNK 256
#define NCHUNK (DN / CHUNK)     // 8
#define SROW (CHUNK + 8)        // 264 halfs = 528B, 16B aligned, bank-spread
#define QROW 24                 // 48B rows, 16B aligned, bank-spread

__global__ __launch_bounds__(128) void otam_wmma_kernel(
    const float* __restrict__ support,  // [Q,S,T,D]
    const float* __restrict__ query,    // [Q,T,D]
    float* __restrict__ out)            // [Q,S]
{
    __shared__ __align__(16) _Float16 sA[TN][SROW];     // support chunk, row-major
    __shared__ __align__(16) _Float16 qT[CHUNK][QROW];  // query chunk, transposed
    __shared__ float red[256];
    __shared__ float inv_ns[TN];
    __shared__ float inv_nq[TN];
    __shared__ float dist[TN][TN + 1];

    const int tid  = threadIdx.x;
    const int lane = tid & 31;
    const int bid  = blockIdx.x;
    const int qi   = bid / SN;
    const int si   = bid % SN;

    const int row = tid >> 3;         // 0..15: frame index this thread stages
    const int seg = (tid & 7) * 32;   // 32-col segment within chunk

    const float* sBase = support + ((size_t)(qi * SN + si) * TN + row) * DN;
    const float* qBase = query   + ((size_t)qi * TN + row) * DN;

    float ss_s = 0.f, ss_q = 0.f;
    v8f acc = {};

    const int offA = (lane < 16) ? 0 : 8;  // A-fragment K sub-offset per ISA layout
    const int rA   = lane & 15;            // A-fragment row (M)

    for (int ch = 0; ch < NCHUNK; ++ch) {
        const int cb = ch * CHUNK;
        if (ch + 1 < NCHUNK) {
            __builtin_prefetch(sBase + cb + CHUNK + seg, 0, 1);
            __builtin_prefetch(qBase + cb + CHUNK + seg, 0, 1);
        }
        // ---- stage support chunk (row-major f16) + norm accumulation ----
        #pragma unroll
        for (int v = 0; v < 8; ++v) {
            const float4 f = *(const float4*)(sBase + cb + seg + v * 4);
            ss_s += f.x * f.x + f.y * f.y + f.z * f.z + f.w * f.w;
            _Float16* d = &sA[row][seg + v * 4];
            d[0] = (_Float16)f.x; d[1] = (_Float16)f.y;
            d[2] = (_Float16)f.z; d[3] = (_Float16)f.w;
        }
        // ---- stage query chunk transposed (B-fragment friendly) ----
        #pragma unroll
        for (int v = 0; v < 8; ++v) {
            const float4 f = *(const float4*)(qBase + cb + seg + v * 4);
            ss_q += f.x * f.x + f.y * f.y + f.z * f.z + f.w * f.w;
            const int k = seg + v * 4;
            qT[k + 0][row] = (_Float16)f.x;
            qT[k + 1][row] = (_Float16)f.y;
            qT[k + 2][row] = (_Float16)f.z;
            qT[k + 3][row] = (_Float16)f.w;
        }
        __syncthreads();

        // ---- WMMA over chunk: acc += S_chunk * Q_chunk^T (all waves, EXEC=all-1s) ----
        #pragma unroll
        for (int ks = 0; ks < CHUNK; ks += 32) {
            v8h alo = *(const v8h*)&sA[rA][ks + offA];
            v8h ahi = *(const v8h*)&sA[rA][ks + offA + 16];
            v16h b;
            #pragma unroll
            for (int s = 0; s < 8; ++s) {
                b[s]     = qT[ks + offA + s][rA];
                b[8 + s] = qT[ks + offA + 16 + s][rA];
            }
            v16h a = __builtin_shufflevector(alo, ahi,
                     0,1,2,3,4,5,6,7,8,9,10,11,12,13,14,15);
            acc = __builtin_amdgcn_wmma_f32_16x16x32_f16(
                      false, a, false, b, (short)0, acc, false, false);
        }
        __syncthreads();
    }

    // ---- row-norm reduction (8 partial sums per row) ----
    red[tid]       = ss_s;
    red[128 + tid] = ss_q;
    __syncthreads();
    if (tid < 16) {
        float s = 0.f;
        #pragma unroll
        for (int k = 0; k < 8; ++k) s += red[tid * 8 + k];
        inv_ns[tid] = 1.f / fmaxf(sqrtf(s), 1e-12f);
    } else if (tid < 32) {
        const int r = tid - 16;
        float s = 0.f;
        #pragma unroll
        for (int k = 0; k < 8; ++k) s += red[128 + r * 8 + k];
        inv_nq[r] = 1.f / fmaxf(sqrtf(s), 1e-12f);
    }
    __syncthreads();

    // ---- wave 0 scatters cosine distance per C/D VGPR layout ----
    if (tid < 32) {
        const int j     = lane & 15;
        const int ibase = (lane < 16) ? 0 : 8;
        #pragma unroll
        for (int r = 0; r < 8; ++r) {
            const int i = ibase + r;
            dist[i][j] = 1.0f - acc[r] * inv_ns[i] * inv_nq[j];
        }
    }
    __syncthreads();

    // ---- OTAM min-plus DP (tiny, fully unrolled in registers) ----
    if (tid == 0) {
        float prev[16], cur[16];
        float run = 0.f;
        #pragma unroll
        for (int j = 0; j < 16; ++j) { run += dist[0][j]; prev[j] = run; }
        #pragma unroll
        for (int i = 1; i < 16; ++i) {
            cur[0] = prev[0] + dist[i][0];
            #pragma unroll
            for (int j = 1; j < 16; ++j)
                cur[j] = dist[i][j] + fminf(prev[j], fminf(cur[j - 1], prev[j - 1]));
            #pragma unroll
            for (int j = 0; j < 16; ++j) prev[j] = cur[j];
        }
        out[qi * SN + si] = prev[15] * (1.0f / 32.0f);
    }
}

extern "C" void kernel_launch(void* const* d_in, const int* in_sizes, int n_in,
                              void* d_out, int out_size, void* d_ws, size_t ws_size,
                              hipStream_t stream) {
    const float* support = (const float*)d_in[0];  // [128,25,16,2048] f32
    const float* query   = (const float*)d_in[1];  // [128,16,2048]    f32
    float* out = (float*)d_out;                    // [128,25]         f32
    (void)in_sizes; (void)n_in; (void)out_size; (void)d_ws; (void)ws_size;

    dim3 grid(QN * SN);   // 3200 workgroups, one per (query, support) pair
    dim3 block(128);      // 4 wave32 waves
    otam_wmma_kernel<<<grid, block, 0, stream>>>(support, query, out);
}
